// MultiHeadAttention_64665027609341
// MI455X (gfx1250) — compile-verified, matrix operand order checked by reading, not run on hardware
//
#include <hip/hip_runtime.h>

// ---------------------------------------------------------------------------
// CDNA5 (gfx1250) wave32 WMMA multi-head attention, async-pipelined.
//   B=2, L=2048, D=1024, H=16, K=64, ROT=32
// Pipeline:
//   1) qkv_rope_kernel : f32 GEMM (double-buffered LDS, f16 WMMA) + fused RoPE
//   2) flash_attn_kernel: streaming softmax(QK^T/8)V; K tiles staged with
//      GLOBAL_LOAD_ASYNC_TO_LDS_B128 (ASYNCcnt), V transposed via VGPR staging
//   3) oproj_kernel    : f16 WMMA GEMM (async A tiles) -> f32 output
// ---------------------------------------------------------------------------

typedef __attribute__((ext_vector_type(16))) _Float16 v16h;
typedef __attribute__((ext_vector_type(8)))  _Float16 v8h;
typedef __attribute__((ext_vector_type(8)))  float    v8f;

#define WMMA_F16(a, b, c) \
  __builtin_amdgcn_wmma_f32_16x16x32_f16(false, (a), false, (b), (short)0, (c), false, false)

// Async memory->LDS copy of 32 contiguous bytes per lane (two B128 ops; the
// 24-bit instruction offset is added to BOTH the LDS and global addresses).
// Tracked by ASYNCcnt; drained with s_wait_asynccnt before the barrier.
#define ASYNC_COPY_32B(ldsaddr, gaddr)                               \
  asm volatile("global_load_async_to_lds_b128 %0, %1, off\n\t"       \
               "global_load_async_to_lds_b128 %0, %1, off offset:16" \
               :: "v"(ldsaddr), "v"(gaddr) : "memory")

#define S_WAIT_ASYNCCNT0() asm volatile("s_wait_asynccnt 0x0" ::: "memory")

static __device__ __forceinline__ unsigned lds_addr_of(const void* p) {
  // Flat LDS addresses carry the LDS byte offset in the low 32 bits
  // (aperture check truncates to the low 32 bits).
  return (unsigned)(unsigned long long)p;
}

static __device__ __forceinline__ v16h frag_cat(v8h lo, v8h hi) {
  return __builtin_shufflevector(lo, hi, 0, 1, 2, 3, 4, 5, 6, 7,
                                          8, 9, 10, 11, 12, 13, 14, 15);
}

// 16x32 f16 fragment (A layout; also B^T layout when the operand is stored
// transposed).  ISA 7.12.2: lane holds row m = lane&15; lanes 0-15 cover
// K = [0..7]+[16..23], lanes 16-31 cover K = [8..15]+[24..31] -> two b128s.
static __device__ __forceinline__ v16h load_frag(const _Float16* __restrict__ base,
                                                 int row0, int col0, int stride) {
  const int lane = threadIdx.x & 31;
  const int m    = lane & 15;
  const int kb   = (lane >> 4) << 3;  // 0 or 8
  const _Float16* p = base + (size_t)(row0 + m) * stride + col0;
  v8h lo = *(const v8h*)(p + kb);
  v8h hi = *(const v8h*)(p + 16 + kb);
  return frag_cat(lo, hi);
}

static __device__ __forceinline__ void zero8(v8f& v) {
#pragma unroll
  for (int i = 0; i < 8; ++i) v[i] = 0.0f;
}

// ---------------------------------------------------------------------------
// Kernel 1: fused QKV projection + RoPE.  grid = (BL/128, H, 3), block = 256.
// Double-buffered LDS; fp32->f16 conversion forces VGPR staging, so the
// "async" here is software pipelining: global loads for tile i+1 issue before
// the WMMAs of tile i; one barrier per iteration.
// ---------------------------------------------------------------------------
__global__ __launch_bounds__(256) void qkv_rope_kernel(
    const float* __restrict__ xq, const float* __restrict__ xk,
    const float* __restrict__ xv, const float* __restrict__ rot,
    const float* __restrict__ wq, const float* __restrict__ wk,
    const float* __restrict__ wv,
    _Float16* __restrict__ oq, _Float16* __restrict__ ok,
    _Float16* __restrict__ ov) {
  constexpr int SA = 40, SB = 40;  // 80B row stride -> conflict-free gathers
  __shared__ _Float16 ldsA[2][128 * SA];
  __shared__ _Float16 ldsB[2][64 * SB];

  const int mode = blockIdx.z;
  const float* __restrict__ X = (mode == 0) ? xq : (mode == 1) ? xk : xv;
  const float* __restrict__ W = (mode == 0) ? wq : (mode == 1) ? wk : wv;
  _Float16* __restrict__ dst  = (mode == 0) ? oq : (mode == 1) ? ok : ov;

  const int mBase = blockIdx.x * 128;
  const int h     = blockIdx.y;
  const int nBase = h * 64;
  const int tid   = threadIdx.x;
  const int w     = tid >> 5;
  const int lane  = tid & 31;

  // Register-staged tiles (software pipeline).
  float aV[16], bV[8];
  const int bD = tid >> 6;      // 0..3
  const int bC = tid & 63;      // 0..63

  auto loadA = [&](int d0) {
#pragma unroll
    for (int i = 0; i < 16; ++i)
      aV[i] = X[(size_t)(mBase + w + 8 * i) * 1024 + d0 + lane];
    if (d0 + 64 < 1024)  // -> global_prefetch_b8, two tiles ahead
      __builtin_prefetch(&X[(size_t)(mBase + w) * 1024 + d0 + 64], 0, 3);
  };
  auto storeA = [&](int buf) {
#pragma unroll
    for (int i = 0; i < 16; ++i)
      ldsA[buf][(w + 8 * i) * SA + lane] = (_Float16)aV[i];
  };
  auto loadB = [&](int d0) {
#pragma unroll
    for (int i = 0; i < 8; ++i)
      bV[i] = W[(size_t)(d0 + bD + 4 * i) * 1024 + nBase + bC];
  };
  auto storeB = [&](int buf) {
#pragma unroll
    for (int i = 0; i < 8; ++i)
      ldsB[buf][bC * SB + bD + 4 * i] = (_Float16)bV[i];
  };

  v8f acc[4];
#pragma unroll
  for (int i = 0; i < 4; ++i) zero8(acc[i]);

  loadA(0); loadB(0);
  storeA(0); storeB(0);
  __syncthreads();

  int cur = 0;
  for (int d0 = 0; d0 < 1024; d0 += 32) {
    const int nxt = cur ^ 1;
    const bool more = (d0 + 32) < 1024;
    if (more) { loadA(d0 + 32); loadB(d0 + 32); }  // overlap with WMMAs

    const v16h a = load_frag(ldsA[cur], w * 16, 0, SA);
#pragma unroll
    for (int nt = 0; nt < 4; ++nt) {
      const v16h b = load_frag(ldsB[cur], nt * 16, 0, SB);
      acc[nt] = WMMA_F16(a, b, acc[nt]);
    }

    if (more) { storeA(nxt); storeB(nxt); }
    __syncthreads();
    cur = nxt;
  }

  // Epilogue: RoPE (q/k, first 32 head dims) + f16 store.  C layout: lane
  // holds column n = lane&15; row = r + 8*(lane>>4); rotate_half partner
  // (n^1) lives in lane^1 -> one shfl_xor.
  const int n  = lane & 15;
  const int hi = lane >> 4;
#pragma unroll
  for (int nt = 0; nt < 4; ++nt) {
    const int k = nt * 16 + n;  // dim within head
#pragma unroll
    for (int r = 0; r < 8; ++r) {
      const int row = mBase + w * 16 + hi * 8 + r;  // in [0, B*L)
      const int l   = row & 2047;
      const int bb  = row >> 11;
      float v = acc[nt][r];
      if (mode < 2 && nt < 2) {  // uniform across the wave
        const float prt = __shfl_xor(v, 1, 32);
        const float f   = rot[l * 32 + k];
        float sf, cf;
        __sincosf(f, &sf, &cf);
        v = (k & 1) ? fmaf(v, cf, prt * sf) : fmaf(v, cf, -prt * sf);
      }
      dst[((size_t)((bb * 16 + h) * 2048 + l)) * 64 + k] = (_Float16)v;
    }
  }
}

// ---------------------------------------------------------------------------
// Kernel 2: flash attention.  grid = (L/128, H, B), block = 256 (8 waves).
// K tiles: GLOBAL_LOAD_ASYNC_TO_LDS_B128 (no VGPR round trip, ASYNCcnt).
// V tiles: VGPR-staged transpose.  Double-buffered; one barrier / key tile.
// ---------------------------------------------------------------------------
__global__ __launch_bounds__(256) void flash_attn_kernel(
    const _Float16* __restrict__ wsQ, const _Float16* __restrict__ wsK,
    const _Float16* __restrict__ wsV, _Float16* __restrict__ wsO) {
  constexpr int S = 72;  // 144B stride -> conflict-free fragment gathers
  __shared__ _Float16 ldsK[2][64 * S];   // K rows: direct B^T for Q.K^T
  __shared__ _Float16 ldsV[2][64 * S];   // V transposed: B^T for P.V
  __shared__ _Float16 ldsP[8 * 16 * S];  // per-wave P scratch (C->A relayout)

  const int h = blockIdx.y, b = blockIdx.z;
  const int tid = threadIdx.x, w = tid >> 5, lane = tid & 31;
  const size_t headOff = (size_t)(b * 16 + h) * 2048 * 64;
  const _Float16* __restrict__ Q  = wsQ + headOff;
  const _Float16* __restrict__ Kg = wsK + headOff;
  const _Float16* __restrict__ Vg = wsV + headOff;

  const int qRow0 = blockIdx.x * 128 + w * 16;
  // Resident Q fragments: 16 rows x 64 dims = two 16x32 A-frags.
  const v16h qf0 = load_frag(Q, qRow0, 0, 64);
  const v16h qf1 = load_frag(Q, qRow0, 32, 64);

  // Staging assignment: thread covers K/V row rS, 16-half segment sS.
  const int rS = tid >> 2, sS = (tid & 3) * 16;
  uint4 vreg[2];

  auto issueK = [&](int kt, int buf) {  // async memory->LDS, 32B per lane
    const unsigned la =
        lds_addr_of(&ldsK[buf][rS * S + sS]);
    const unsigned long long ga =
        (unsigned long long)(Kg + (size_t)(kt + rS) * 64 + sS);
    ASYNC_COPY_32B(la, ga);
  };
  auto loadV = [&](int kt) {
    const uint4* src = (const uint4*)(Vg + (size_t)(kt + rS) * 64 + sS);
    vreg[0] = src[0];
    vreg[1] = src[1];
  };
  auto storeV = [&](int buf) {  // transpose to ldsV[d][key]
    const _Float16* hv = (const _Float16*)vreg;
#pragma unroll
    for (int j = 0; j < 16; ++j) ldsV[buf][(sS + j) * S + rS] = hv[j];
  };

  v8f o[4];
  float mi[8], li[8];
#pragma unroll
  for (int i = 0; i < 4; ++i) zero8(o[i]);
#pragma unroll
  for (int r = 0; r < 8; ++r) { mi[r] = -__builtin_inff(); li[r] = 0.0f; }

  _Float16* __restrict__ myP = ldsP + w * 16 * S;
  const int n = lane & 15, hiL = lane >> 4;

  issueK(0, 0);
  loadV(0);
  storeV(0);
  S_WAIT_ASYNCCNT0();
  __syncthreads();

  int cur = 0;
  for (int kt = 0; kt < 2048; kt += 64) {
    const int nxt = cur ^ 1;
    const bool more = (kt + 64) < 2048;
    if (more) { issueK(kt + 64, nxt); loadV(kt + 64); }  // overlap w/ compute

    // ---- S = Q x K^T : 16x64 in 4 accumulators -------------------------
    v8f sc[4];
#pragma unroll
    for (int i = 0; i < 4; ++i) zero8(sc[i]);
#pragma unroll
    for (int nt = 0; nt < 4; ++nt) {
      const v16h b0 = load_frag(ldsK[cur], nt * 16, 0, S);
      const v16h b1 = load_frag(ldsK[cur], nt * 16, 32, S);
      sc[nt] = WMMA_F16(qf0, b0, sc[nt]);
      sc[nt] = WMMA_F16(qf1, b1, sc[nt]);
    }

    // ---- online softmax (row stats per register; half-wave reductions) --
#pragma unroll
    for (int r = 0; r < 8; ++r) {
      const float s0 = sc[0][r] * 0.125f;  // 1/sqrt(64)
      const float s1 = sc[1][r] * 0.125f;
      const float s2 = sc[2][r] * 0.125f;
      const float s3 = sc[3][r] * 0.125f;
      float mx = fmaxf(fmaxf(s0, s1), fmaxf(s2, s3));
      mx = fmaxf(mx, __shfl_xor(mx, 1, 32));
      mx = fmaxf(mx, __shfl_xor(mx, 2, 32));
      mx = fmaxf(mx, __shfl_xor(mx, 4, 32));
      mx = fmaxf(mx, __shfl_xor(mx, 8, 32));
      const float mnew  = fmaxf(mi[r], mx);
      const float alpha = __expf(mi[r] - mnew);
      const float p0 = __expf(s0 - mnew), p1 = __expf(s1 - mnew);
      const float p2 = __expf(s2 - mnew), p3 = __expf(s3 - mnew);
      const int pr = (r + hiL * 8) * S + n;
      myP[pr]      = (_Float16)p0;
      myP[pr + 16] = (_Float16)p1;
      myP[pr + 32] = (_Float16)p2;
      myP[pr + 48] = (_Float16)p3;
      float ps = p0 + p1 + p2 + p3;
      ps += __shfl_xor(ps, 1, 32);
      ps += __shfl_xor(ps, 2, 32);
      ps += __shfl_xor(ps, 4, 32);
      ps += __shfl_xor(ps, 8, 32);
      li[r] = li[r] * alpha + ps;
      mi[r] = mnew;
#pragma unroll
      for (int dt = 0; dt < 4; ++dt) o[dt][r] *= alpha;
    }

    // ---- O += P x V (P re-read from LDS in A layout; same-wave in-order) -
    const v16h p0f = load_frag(myP, 0, 0, S);
    const v16h p1f = load_frag(myP, 0, 32, S);
#pragma unroll
    for (int dt = 0; dt < 4; ++dt) {
      const v16h b0 = load_frag(ldsV[cur], dt * 16, 0, S);
      const v16h b1 = load_frag(ldsV[cur], dt * 16, 32, S);
      o[dt] = WMMA_F16(p0f, b0, o[dt]);
      o[dt] = WMMA_F16(p1f, b1, o[dt]);
    }

    if (more) storeV(nxt);  // ds stores into next buffer (no readers yet)
    S_WAIT_ASYNCCNT0();     // own async K-tile writes complete
    __syncthreads();        // all waves' tile writes visible
    cur = nxt;
  }

  // Normalize and store as [(b*L+l)][h*64+d] f16 (feeds output projection).
#pragma unroll
  for (int dt = 0; dt < 4; ++dt) {
#pragma unroll
    for (int r = 0; r < 8; ++r) {
      const int l = qRow0 + hiL * 8 + r;
      const float v = o[dt][r] / li[r];
      wsO[(size_t)(b * 2048 + l) * 1024 + h * 64 + dt * 16 + n] = (_Float16)v;
    }
  }
}

// ---------------------------------------------------------------------------
// Kernel 3: output projection.  grid = (BL/128, D/64), block = 256.
// A tiles (f16) staged with async-to-LDS; W tiles fp32->f16 via registers.
// ---------------------------------------------------------------------------
__global__ __launch_bounds__(256) void oproj_kernel(
    const _Float16* __restrict__ A, const float* __restrict__ Wo,
    float* __restrict__ out) {
  constexpr int SA = 40, SB = 40;
  __shared__ _Float16 ldsA[2][128 * SA];
  __shared__ _Float16 ldsB[2][64 * SB];

  const int mBase = blockIdx.x * 128;
  const int nBase = blockIdx.y * 64;
  const int tid = threadIdx.x, w = tid >> 5, lane = tid & 31;

  const int rA = tid >> 1, sA = (tid & 1) * 16;  // A: 128 rows x 2 segments
  const int bD = tid >> 6, bC = tid & 63;
  float bV[8];

  auto issueA = [&](int d0, int buf) {
    const unsigned la = lds_addr_of(&ldsA[buf][rA * SA + sA]);
    const unsigned long long ga =
        (unsigned long long)(A + (size_t)(mBase + rA) * 1024 + d0 + sA);
    ASYNC_COPY_32B(la, ga);
  };
  auto loadB = [&](int d0) {
#pragma unroll
    for (int i = 0; i < 8; ++i)
      bV[i] = Wo[(size_t)(d0 + bD + 4 * i) * 1024 + nBase + bC];
    if (d0 + 64 < 1024)
      __builtin_prefetch(&Wo[(size_t)(d0 + 64 + bD) * 1024 + nBase + bC], 0, 3);
  };
  auto storeB = [&](int buf) {
#pragma unroll
    for (int i = 0; i < 8; ++i)
      ldsB[buf][bC * SB + bD + 4 * i] = (_Float16)bV[i];
  };

  v8f acc[4];
#pragma unroll
  for (int i = 0; i < 4; ++i) zero8(acc[i]);

  issueA(0, 0);
  loadB(0);
  storeB(0);
  S_WAIT_ASYNCCNT0();
  __syncthreads();

  int cur = 0;
  for (int d0 = 0; d0 < 1024; d0 += 32) {
    const int nxt = cur ^ 1;
    const bool more = (d0 + 32) < 1024;
    if (more) { issueA(d0 + 32, nxt); loadB(d0 + 32); }

    const v16h a = load_frag(ldsA[cur], w * 16, 0, SA);
#pragma unroll
    for (int nt = 0; nt < 4; ++nt) {
      const v16h b = load_frag(ldsB[cur], nt * 16, 0, SB);
      acc[nt] = WMMA_F16(a, b, acc[nt]);
    }

    if (more) storeB(nxt);
    S_WAIT_ASYNCCNT0();
    __syncthreads();
    cur = nxt;
  }

  const int n = lane & 15, hi = lane >> 4;
#pragma unroll
  for (int nt = 0; nt < 4; ++nt) {
#pragma unroll
    for (int r = 0; r < 8; ++r) {
      out[(size_t)(mBase + w * 16 + hi * 8 + r) * 1024 + nBase + nt * 16 + n] =
          acc[nt][r];
    }
  }
}

// ---------------------------------------------------------------------------
extern "C" void kernel_launch(void* const* d_in, const int* in_sizes, int n_in,
                              void* d_out, int out_size, void* d_ws,
                              size_t ws_size, hipStream_t stream) {
  (void)in_sizes; (void)n_in; (void)out_size; (void)ws_size;
  const float* query = (const float*)d_in[0];
  const float* key   = (const float*)d_in[1];
  const float* value = (const float*)d_in[2];
  const float* rot   = (const float*)d_in[3];
  const float* qw    = (const float*)d_in[4];
  const float* kw    = (const float*)d_in[5];
  const float* vw    = (const float*)d_in[6];
  const float* ow    = (const float*)d_in[7];
  float* out = (float*)d_out;

  const size_t perMat = (size_t)2 * 16 * 2048 * 64;  // B*H*L*K halves
  _Float16* wsQ = (_Float16*)d_ws;
  _Float16* wsK = wsQ + perMat;
  _Float16* wsV = wsK + perMat;
  _Float16* wsO = wsV + perMat;  // [(b*L+l)][H*64] halves

  const dim3 blk(256);
  qkv_rope_kernel<<<dim3(32, 16, 3), blk, 0, stream>>>(
      query, key, value, rot, qw, kw, vw, wsQ, wsK, wsV);
  flash_attn_kernel<<<dim3(16, 16, 2), blk, 0, stream>>>(wsQ, wsK, wsV, wsO);
  oproj_kernel<<<dim3(32, 16, 1), blk, 0, stream>>>(wsO, ow, out);
}
